// AttentionLSTM_60550448939165
// MI455X (gfx1250) — compile-verified, matrix operand order checked by reading, not run on hardware
//
#include <hip/hip_runtime.h>

typedef __attribute__((ext_vector_type(16))) __bf16 v16bf;
typedef __attribute__((ext_vector_type(8)))  float  v8f;

#define N_B   64
#define T_S   256
#define D_IN  1024
#define H_DIM 1024
#define G4    4096   // 4*H

// ---------- bf16 helpers (bit-exact, no reliance on __bf16 arithmetic) ----------
__device__ __forceinline__ __bf16 f2bf(float f) {
    union { float f; unsigned u; } x; x.f = f;
    unsigned r = (x.u + 0x7FFFu + ((x.u >> 16) & 1u)) >> 16;   // round-to-nearest-even
    union { unsigned short s; __bf16 b; } y; y.s = (unsigned short)r;
    return y.b;
}
__device__ __forceinline__ float bf2f(__bf16 b) {
    union { __bf16 b; unsigned short s; } x; x.b = b;
    union { unsigned u; float f; } y; y.u = ((unsigned)x.s) << 16;
    return y.f;
}

// Load one 16-bf16 WMMA fragment as two contiguous 16-byte chunks.
__device__ __forceinline__ v16bf load_frag(const __bf16* p0, const __bf16* p1) {
    union { v16bf v; uint4 u[2]; } f;
    f.u[0] = *reinterpret_cast<const uint4*>(p0);
    f.u[1] = *reinterpret_cast<const uint4*>(p1);
    return f.v;
}

// ---------- prep kernels ----------
__global__ void k_cvt_x(const float* __restrict__ x, __bf16* __restrict__ xb, int n) {
    int i = blockIdx.x * blockDim.x + threadIdx.x;
    if (i < n) xb[i] = f2bf(x[i]);
}

// Build transposed bf16 weights: WxT[col][k] = Wx[k][col], WhT[col][k] = Wh[k][col]
__global__ void k_build_wt(const float* __restrict__ Wx, const float* __restrict__ Wh,
                           __bf16* __restrict__ wxT, __bf16* __restrict__ whT) {
    int i = blockIdx.x * blockDim.x + threadIdx.x;   // 4096 * 2048 threads
    int col = i & (G4 - 1);
    int k   = i >> 12;
    if (k < D_IN) wxT[(size_t)col * D_IN + k] = f2bf(Wx[(size_t)k * G4 + col]);
    else          whT[(size_t)col * H_DIM + (k - D_IN)] = f2bf(Wh[(size_t)(k - D_IN) * G4 + col]);
}

// h0 = c0 = mean(A over P*P=100); write f32 c and bf16 h0
__global__ void k_h0(const float* __restrict__ A, float* __restrict__ c,
                     __bf16* __restrict__ h0) {
    int i = blockIdx.x * blockDim.x + threadIdx.x;   // N*H = 65536
    const float* p = A + (size_t)i * 100;
    float s = 0.f;
    #pragma unroll 4
    for (int j = 0; j < 100; ++j) s += p[j];
    float m = s * 0.01f;
    c[i]  = m;
    h0[i] = f2bf(m);
}

// ---------- big parallel GEMM: XW = x @ Wx + b  (bf16 out) ----------
__global__ void __launch_bounds__(128)
k_gemm_xw(const __bf16* __restrict__ xb, const __bf16* __restrict__ wxT,
          const float* __restrict__ bias, __bf16* __restrict__ xw) {
    const int lane = threadIdx.x & 31;
    const int wv   = threadIdx.x >> 5;
    const int half = lane >> 4;
    const int l15  = lane & 15;
    const int row0 = blockIdx.y * 16;                 // 0..16383 rows (N*T)
    const int col0 = (blockIdx.x * 4 + wv) * 16;      // 0..4095 cols (4H)

    const __bf16* Ab = xb  + (size_t)(row0 + l15) * D_IN + half * 8;
    const __bf16* Bb = wxT + (size_t)(col0 + l15) * D_IN + half * 8;

    v8f acc = {0.f,0.f,0.f,0.f,0.f,0.f,0.f,0.f};
    for (int kb = 0; kb < D_IN; kb += 32) {
        __builtin_prefetch(Bb + kb + 128, 0, 1);      // global_prefetch_b8 on B stream
        v16bf a = load_frag(Ab + kb, Ab + kb + 16);
        v16bf b = load_frag(Bb + kb, Bb + kb + 16);
        acc = __builtin_amdgcn_wmma_f32_16x16x32_bf16(false, a, false, b,
                                                      (short)0, acc, false, false);
    }
    #pragma unroll
    for (int r = 0; r < 8; ++r) {
        int rr = row0 + r + 8 * half;     // C/D layout: VGPR r -> M = r + 8*(lane>=16)
        int cc = col0 + l15;              // N = lane & 15
        xw[(size_t)rr * G4 + cc] = f2bf(acc[r] + bias[cc]);
    }
}

// ---------- one recurrence step: a = XW[:,t,:] + h_prev @ Wh; gates; c/h update ----------
__global__ void __launch_bounds__(128)
k_step(const __bf16* __restrict__ hb_prev, __bf16* __restrict__ hb_next,
       const __bf16* __restrict__ whT, const __bf16* __restrict__ xw,
       float* __restrict__ c, float* __restrict__ out, int t) {
    __shared__ float gates[4][16][16];

    const int lane  = threadIdx.x & 31;
    const int g     = threadIdx.x >> 5;          // wave id = gate (i,f,o,g)
    const int half  = lane >> 4;
    const int l15   = lane & 15;
    const int row0  = blockIdx.y * 16;           // batch tile (N=64 -> 4 tiles)
    const int col0  = blockIdx.x * 16;           // hidden tile (H=1024 -> 64 tiles)
    const int gcol0 = g * H_DIM + col0;          // column in 4H gate space

    // Seed accumulator with precomputed x@Wx + b tile (bf16 -> f32)
    v8f acc;
    #pragma unroll
    for (int r = 0; r < 8; ++r) {
        int rr = row0 + r + 8 * half;            // batch row
        acc[r] = bf2f(xw[((size_t)rr * T_S + t) * G4 + gcol0 + l15]);
    }

    const __bf16* Ab = hb_prev + (size_t)(row0 + l15) * H_DIM + half * 8;
    const __bf16* Bb = whT + (size_t)(gcol0 + l15) * H_DIM + half * 8;
    for (int kb = 0; kb < H_DIM; kb += 32) {
        v16bf a = load_frag(Ab + kb, Ab + kb + 16);
        v16bf b = load_frag(Bb + kb, Bb + kb + 16);
        acc = __builtin_amdgcn_wmma_f32_16x16x32_bf16(false, a, false, b,
                                                      (short)0, acc, false, false);
    }

    #pragma unroll
    for (int r = 0; r < 8; ++r) gates[g][r + 8 * half][l15] = acc[r];
    __syncthreads();

    // Elementwise cell update: 256 elements, 128 threads
    for (int e = threadIdx.x; e < 256; e += 128) {
        int m    = e >> 4;
        int nn   = e & 15;
        int n_b  = row0 + m;
        int hcol = col0 + nn;
        float ai = gates[0][m][nn];
        float af = gates[1][m][nn];
        float ao = gates[2][m][nn];
        float ag = gates[3][m][nn];
        float i_ = 1.f / (1.f + __expf(-ai));
        float f_ = 1.f / (1.f + __expf(-af));
        float o_ = 1.f / (1.f + __expf(-ao));
        float g_ = tanhf(ag);
        size_t ci = (size_t)n_b * H_DIM + hcol;
        float cn = f_ * c[ci] + i_ * g_;
        float hn = o_ * tanhf(cn);
        c[ci]       = cn;
        hb_next[ci] = f2bf(hn);
        out[((size_t)n_b * T_S + t) * H_DIM + hcol] = hn;
    }
}

extern "C" void kernel_launch(void* const* d_in, const int* in_sizes, int n_in,
                              void* d_out, int out_size, void* d_ws, size_t ws_size,
                              hipStream_t stream) {
    const float* x  = (const float*)d_in[0];
    const float* A  = (const float*)d_in[1];
    const float* Wx = (const float*)d_in[2];
    const float* Wh = (const float*)d_in[3];
    const float* b  = (const float*)d_in[4];
    float* out = (float*)d_out;

    // Workspace carve-out (~185 MB total), 256B aligned slices
    char* ws = (char*)d_ws;
    size_t off = 0;
    auto carve = [&](size_t bytes) -> void* {
        void* p = ws + off;
        off = (off + bytes + 255) & ~(size_t)255;
        return p;
    };
    __bf16* xb  = (__bf16*)carve((size_t)N_B * T_S * D_IN * 2);   // 33.5 MB
    __bf16* wxT = (__bf16*)carve((size_t)G4 * D_IN * 2);          //  8.4 MB
    __bf16* whT = (__bf16*)carve((size_t)G4 * H_DIM * 2);         //  8.4 MB (L2-resident)
    __bf16* xw  = (__bf16*)carve((size_t)N_B * T_S * G4 * 2);     // 134 MB
    float*  c   = (float*) carve((size_t)N_B * H_DIM * 4);        // cell state f32
    __bf16* hb0 = (__bf16*)carve((size_t)N_B * H_DIM * 2);        // h ping
    __bf16* hb1 = (__bf16*)carve((size_t)N_B * H_DIM * 2);        // h pong

    // Prep: convert x, transpose weights, compute h0/c0
    k_cvt_x  <<<(N_B * T_S * D_IN) / 256, 256, 0, stream>>>(x, xb, N_B * T_S * D_IN);
    k_build_wt<<<(G4 * (D_IN + H_DIM)) / 256, 256, 0, stream>>>(Wx, Wh, wxT, whT);
    k_h0     <<<(N_B * H_DIM) / 256, 256, 0, stream>>>(A, c, hb0);

    // Fully parallel XW = x @ Wx + b
    k_gemm_xw<<<dim3(G4 / 16 / 4, (N_B * T_S) / 16), 128, 0, stream>>>(xb, wxT, b, xw);

    // Sequential recurrence: 256 step kernels, h ping-pong
    __bf16* bufs[2] = { hb0, hb1 };
    for (int t = 0; t < T_S; ++t) {
        k_step<<<dim3(H_DIM / 16, N_B / 16), 128, 0, stream>>>(
            bufs[t & 1], bufs[(t + 1) & 1], whT, xw, c, out, t);
    }
}
